// QwenTorchMultiHeadAttentionHandler_63050119905466
// MI455X (gfx1250) — matrix operand order repaired and checked by reading, not compiled, not run on hardware
//
#include <hip/hip_runtime.h>

typedef __attribute__((ext_vector_type(16))) _Float16 v16h;
typedef __attribute__((ext_vector_type(8)))  _Float16 v8h;
typedef __attribute__((ext_vector_type(4)))  _Float16 v4h;
typedef __attribute__((ext_vector_type(8)))  float    v8f;

#define SEQ 4096
#define NH 16
#define HD 80
#define HDP 96             // head dim padded to 3 x 32 for 16x16x32 WMMA
#define ROW_STRIDE (NH*HD) // 1280
#define QT 128             // q rows per workgroup (8 waves x 16 rows)
#define KT 64              // kv cols per tile
#define NTHREADS 256

// (1/sqrt(80)) * log2(e): fold softmax scale AND exp->exp2 base change into Q
#define SCALE_LOG2E 0.16129841386f

#if __has_builtin(__builtin_amdgcn_exp2f)
#define EXP2F(x) __builtin_amdgcn_exp2f(x)      // raw v_exp_f32
#else
#define EXP2F(x) __expf((x) * 0.6931471805599453f)
#endif

__launch_bounds__(NTHREADS, 1)
__global__ void fa_varlen_wmma_kernel(const float* __restrict__ Q,
                                      const float* __restrict__ K,
                                      const float* __restrict__ V,
                                      const int* __restrict__ cu,
                                      int n_cu,
                                      float* __restrict__ Out)
{
    __shared__ _Float16 Qs[QT][HDP];      // 24 KB (pre-scaled by SCALE_LOG2E)
    __shared__ _Float16 Ks[KT][HDP];      // 12 KB
    __shared__ _Float16 Vt[HD][KT];       // 10 KB, transposed V tile
    __shared__ _Float16 Ps[QT][KT];       // 16 KB probabilities (f16)

    const int tid  = threadIdx.x;
    const int wave = tid >> 5;
    const int lane = tid & 31;
    const int l16  = lane & 15;
    const int hi   = (lane >> 4) & 1;
    const int h    = blockIdx.y;
    const int q0   = blockIdx.x * QT;
    const int rowbase = wave * 16;

    // segment bounds for this q-tile from cu_seqlens (block-diagonal mask)
    int kv_start = 0, kv_end = SEQ;
    for (int s = 0; s + 1 < n_cu; ++s) {
        int a = cu[s], b = cu[s + 1];
        if (q0 >= a && q0 < b) { kv_start = a; kv_end = b; }
    }
    const int ntile = (kv_end - kv_start + KT - 1) / KT;

    // ---- stage Q tile: 128 x 80 f32 -> f16 * SCALE_LOG2E, zero-pad to 96 ----
    for (int i = 0; i < 10; ++i) {
        int idx = tid + i * NTHREADS;            // 128 rows * 20 float4
        int r = idx / 20, c4 = idx % 20;
        const float4 f = *(const float4*)(Q + (size_t)(q0 + r) * ROW_STRIDE + h * HD + c4 * 4);
        v4h hv;
        hv[0] = (_Float16)(f.x * SCALE_LOG2E);
        hv[1] = (_Float16)(f.y * SCALE_LOG2E);
        hv[2] = (_Float16)(f.z * SCALE_LOG2E);
        hv[3] = (_Float16)(f.w * SCALE_LOG2E);
        *(v4h*)&Qs[r][c4 * 4] = hv;
    }
    { int r = tid >> 1, part = tid & 1; v8h z = {}; *(v8h*)&Qs[r][HD + part * 8] = z; }
    __syncthreads();

    // per-lane running stats for the 8 rows this lane's C/O fragments hold;
    // l_run holds a per-lane PARTIAL sum (reduced across lanes only at the end)
    float m_run[8], l_run[8];
    #pragma unroll
    for (int r = 0; r < 8; ++r) { m_run[r] = -__builtin_inff(); l_run[r] = 0.f; }
    v8f Of[5];
    #pragma unroll
    for (int nt = 0; nt < 5; ++nt) { v8f z = {}; Of[nt] = z; }

    for (int t = 0; t < ntile; ++t) {
        const int kvt0 = kv_start + t * KT;
        __syncthreads();                        // previous tile fully consumed

        // ---- stage K tile (row-major f16, padded): 64 rows * 20 float4 ----
        for (int i = 0; i < 5; ++i) {
            int idx = tid + i * NTHREADS;
            int r = idx / 20, c4 = idx % 20;
            v4h hv = {};
            if (kvt0 + r < kv_end) {
                const float4 f = *(const float4*)(K + (size_t)(kvt0 + r) * ROW_STRIDE + h * HD + c4 * 4);
                hv[0] = (_Float16)f.x; hv[1] = (_Float16)f.y;
                hv[2] = (_Float16)f.z; hv[3] = (_Float16)f.w;
            }
            *(v4h*)&Ks[r][c4 * 4] = hv;
        }
        if (tid < 2 * KT) { int r = tid >> 1, part = tid & 1; v8h z = {}; *(v8h*)&Ks[r][HD + part * 8] = z; }

        // ---- stage V tile transposed: Vt[col][kvrow] ----
        for (int i = 0; i < 5; ++i) {
            int idx = tid + i * NTHREADS;
            int r = idx / 20, c4 = idx % 20;
            float4 f = make_float4(0.f, 0.f, 0.f, 0.f);
            if (kvt0 + r < kv_end)
                f = *(const float4*)(V + (size_t)(kvt0 + r) * ROW_STRIDE + h * HD + c4 * 4);
            Vt[c4 * 4 + 0][r] = (_Float16)f.x;
            Vt[c4 * 4 + 1][r] = (_Float16)f.y;
            Vt[c4 * 4 + 2][r] = (_Float16)f.z;
            Vt[c4 * 4 + 3][r] = (_Float16)f.w;
        }
        __syncthreads();

        // ---- prefetch next KV tile into cache while we compute this one ----
        if (t + 1 < ntile) {
            const int kvn = kvt0 + KT;
            for (int i = 0; i < 5; ++i) {
                int idx = tid + i * NTHREADS;
                int r = idx / 20, c4 = idx % 20;
                if ((c4 & 1) == 0 && kvn + r < kv_end) {   // one per 32B (every other float4)
                    __builtin_prefetch(K + (size_t)(kvn + r) * ROW_STRIDE + h * HD + c4 * 4, 0, 3);
                    __builtin_prefetch(V + (size_t)(kvn + r) * ROW_STRIDE + h * HD + c4 * 4, 0, 3);
                }
            }
        }

        // ---- S = (Q*scale) * K^T : 4 N-tiles x 3 K-chunks of 32 ----
        v8f acc[4];
        #pragma unroll
        for (int nt = 0; nt < 4; ++nt) { v8f z = {}; acc[nt] = z; }
        #pragma unroll
        for (int kc = 0; kc < 3; ++kc) {
            const int kb = kc * 32;
            union { v16h v; v8h h2[2]; } a;
            a.h2[0] = *(const v8h*)&Qs[rowbase + l16][kb + 16 * hi];
            a.h2[1] = *(const v8h*)&Qs[rowbase + l16][kb + 16 * hi + 8];
            #pragma unroll
            for (int nt = 0; nt < 4; ++nt) {
                const v16h b = *(const v16h*)&Ks[nt * 16 + l16][kb + 16 * hi];
                acc[nt] = __builtin_amdgcn_wmma_f32_16x16x32_f16(
                    false, a.v, false, b, (short)0, acc[nt], false, false);
            }
        }

        // column-valid masks (block-diagonal tail guard)
        bool valid[4];
        #pragma unroll
        for (int nt = 0; nt < 4; ++nt) valid[nt] = (kvt0 + nt * 16 + l16) < kv_end;

        // ---- in-register online softmax (base-2 domain) ----
        // row max: each row's 16 columns live in one 16-lane half -> xor reduce
        float mt[8];
        #pragma unroll
        for (int r = 0; r < 8; ++r) {
            float v0 = valid[0] ? acc[0][r] : -__builtin_inff();
            float v1 = valid[1] ? acc[1][r] : -__builtin_inff();
            float v2 = valid[2] ? acc[2][r] : -__builtin_inff();
            float v3 = valid[3] ? acc[3][r] : -__builtin_inff();
            mt[r] = fmaxf(fmaxf(v0, v1), fmaxf(v2, v3));
        }
        #pragma unroll
        for (int d = 1; d < 16; d <<= 1)
            #pragma unroll
            for (int r = 0; r < 8; ++r)
                mt[r] = fmaxf(mt[r], __shfl_xor(mt[r], d));

        float alpha[8];
        #pragma unroll
        for (int r = 0; r < 8; ++r) {
            const float m_new = fmaxf(m_run[r], mt[r]);
            alpha[r] = EXP2F(m_run[r] - m_new);
            m_run[r] = m_new;
        }
        // p, partial row-sum (NOT lane-reduced: recurrence is linear in l),
        // and C->A transpose of P via LDS
        #pragma unroll
        for (int r = 0; r < 8; ++r) {
            float ps = 0.f;
            #pragma unroll
            for (int nt = 0; nt < 4; ++nt) {
                const float p = valid[nt] ? EXP2F(acc[nt][r] - m_run[r]) : 0.f;
                ps += p;
                Ps[rowbase + r + 8 * hi][nt * 16 + l16] = (_Float16)p;
            }
            l_run[r] = l_run[r] * alpha[r] + ps;
        }

        // ---- rescale O (rows match this lane's alpha), then O += P * V ----
        #pragma unroll
        for (int r = 0; r < 8; ++r)
            #pragma unroll
            for (int nt = 0; nt < 5; ++nt) Of[nt][r] *= alpha[r];

        union { v16h v; v8h h2[2]; } pa[2];
        #pragma unroll
        for (int kc = 0; kc < 2; ++kc) {
            const int kb = kc * 32;
            pa[kc].h2[0] = *(const v8h*)&Ps[rowbase + l16][kb + 16 * hi];
            pa[kc].h2[1] = *(const v8h*)&Ps[rowbase + l16][kb + 16 * hi + 8];
        }
        #pragma unroll
        for (int nt = 0; nt < 5; ++nt) {
            #pragma unroll
            for (int kc = 0; kc < 2; ++kc) {
                const v16h b = *(const v16h*)&Vt[nt * 16 + l16][kc * 32 + 16 * hi];
                Of[nt] = __builtin_amdgcn_wmma_f32_16x16x32_f16(
                    false, pa[kc].v, false, b, (short)0, Of[nt], false, false);
            }
        }
    }

    // ---- one-time lane reduction of the partial l sums, then store ----
    #pragma unroll
    for (int d = 1; d < 16; d <<= 1)
        #pragma unroll
        for (int r = 0; r < 8; ++r)
            l_run[r] += __shfl_xor(l_run[r], d);

    #pragma unroll
    for (int r = 0; r < 8; ++r) {
        const float li = 1.0f / l_run[r];
        const int row = q0 + rowbase + r + 8 * hi;
        #pragma unroll
        for (int nt = 0; nt < 5; ++nt)
            Out[(size_t)row * ROW_STRIDE + h * HD + nt * 16 + l16] = Of[nt][r] * li;
    }
}

extern "C" void kernel_launch(void* const* d_in, const int* in_sizes, int n_in,
                              void* d_out, int out_size, void* d_ws, size_t ws_size,
                              hipStream_t stream) {
    const float* q  = (const float*)d_in[0];
    const float* k  = (const float*)d_in[1];
    const float* v  = (const float*)d_in[2];
    // d_in[3] = seq_length (scalar, compile-time constant here)
    const int*   cu = (const int*)d_in[4];
    const int n_cu  = in_sizes[4];

    dim3 grid(SEQ / QT, NH);
    fa_varlen_wmma_kernel<<<grid, NTHREADS, 0, stream>>>(q, k, v, cu, n_cu, (float*)d_out);
}
